// Coordinator_15049565405873
// MI455X (gfx1250) — matrix operand order repaired and checked by reading, not compiled, not run on hardware
//
#include <hip/hip_runtime.h>
#include <hip/hip_bf16.h>

// ---------------- CDNA5 (gfx1250) bf16 WMMA bidirectional-GRU coordinator ----------------
#define NA 8      // agents (== seq length T)
#define NB 4096   // batch
#define NP 128    // plan size
#define NH 256    // hidden
#define NT 8      // timesteps (== NA)

typedef __attribute__((ext_vector_type(16))) __bf16 v16bf;
typedef __attribute__((ext_vector_type(8)))  float  v8f;

union FragA { unsigned short us[16]; uint4 q[2]; v16bf v; };

__device__ __forceinline__ unsigned short f2bf_u(float f) {
  unsigned int u = __float_as_uint(f);
  u += 0x7FFFu + ((u >> 16) & 1u);     // round-to-nearest-even to bf16
  return (unsigned short)(u >> 16);
}

// ---------------------------------------------------------------------------
// f32 -> bf16 bulk converter (one-time prepack; bandwidth-bound)
__global__ __launch_bounds__(256) void cvt_bf16_kernel(
    const float* __restrict__ in, unsigned short* __restrict__ out, int n) {
  int i = blockIdx.x * 1024 + threadIdx.x * 4;
  if (i + 3 < n) {
    float4 f = *(const float4*)(in + i);
    ushort4 o;
    o.x = f2bf_u(f.x); o.y = f2bf_u(f.y); o.z = f2bf_u(f.z); o.w = f2bf_u(f.w);
    *(ushort4*)(out + i) = o;
  } else {
    for (int k = i; k < n; ++k) out[k] = f2bf_u(in[k]);
  }
}

// ---------------------------------------------------------------------------
// Kernel 1: per-(t,b) nonzero flags for comm_plans and plans. One wave per item.
__global__ __launch_bounds__(256) void mask_kernel(
    const float* __restrict__ plans, const float* __restrict__ comm_plans,
    unsigned char* __restrict__ cp_nz, unsigned char* __restrict__ pl_nz) {
  __shared__ int fc[8], fp[8];
  const int wave = threadIdx.x >> 5, lane = threadIdx.x & 31;
  if (threadIdx.x < 8) { fc[threadIdx.x] = 0; fp[threadIdx.x] = 0; }
  __syncthreads();
  const int item = blockIdx.x * 8 + wave;           // t*NB + b
  const int t = item >> 12, b = item & (NB - 1);
  const size_t base = ((size_t)b * NA + t) * NP + lane * 4;
  float4 c4 = *(const float4*)(comm_plans + base);
  float4 p4 = *(const float4*)(plans + base);
  if (c4.x != 0.f || c4.y != 0.f || c4.z != 0.f || c4.w != 0.f) fc[wave] = 1;
  if (p4.x != 0.f || p4.y != 0.f || p4.z != 0.f || p4.w != 0.f) fp[wave] = 1;
  __syncthreads();
  if (lane == 0) {
    cp_nz[item] = (unsigned char)fc[wave];
    pl_nz[item] = (unsigned char)fp[wave];
  }
}

// ---------------------------------------------------------------------------
// Kernel 2 (single block, 1024 thr): row_has, mask, lengths, compaction order,
// and the masked_scatter prefix map  srcMap[t*NB+b] = exclusive prefix count (or -1).
__global__ __launch_bounds__(1024) void scan_kernel(
    const unsigned char* __restrict__ cp_nz, const unsigned char* __restrict__ pl_nz,
    unsigned char* __restrict__ mask, unsigned char* __restrict__ row_has,
    int* __restrict__ lenb, unsigned char* __restrict__ order_valid,
    int* __restrict__ srcMap) {
  const int tid = threadIdx.x;
  for (int b = tid; b < NB; b += 1024) {
    unsigned char rh = 0;
    for (int t = 0; t < NT; ++t) rh |= cp_nz[t * NB + b];
    row_has[b] = rh ? 1 : 0;
    for (int t = 0; t < NT; ++t) order_valid[t * NB + b] = 0;
    int len = 0;
    for (int t = 0; t < NT; ++t) {
      unsigned char m = rh ? cp_nz[t * NB + b] : pl_nz[t * NB + b];
      mask[t * NB + b] = m;
      if (m) { order_valid[len * NB + b] = (unsigned char)t; ++len; }
    }
    lenb[b] = len;
  }
  __syncthreads();
  __shared__ int ssum[1024];
  const int base = tid * 32;
  unsigned char loc[32];
  int s = 0;
  for (int e = 0; e < 32; ++e) { loc[e] = mask[base + e]; s += loc[e]; }
  ssum[tid] = s;
  __syncthreads();
  for (int off = 1; off < 1024; off <<= 1) {
    int v = (tid >= off) ? ssum[tid - off] : 0;
    __syncthreads();
    ssum[tid] += v;
    __syncthreads();
  }
  int run = ssum[tid] - s;                 // exclusive prefix at base
  for (int e = 0; e < 32; ++e) {
    srcMap[base + e] = loc[e] ? run : -1;
    run += loc[e];
  }
}

// ---------------------------------------------------------------------------
// Kernel 3: one fused GRU step for one agent, both directions (grid.y).
// GEMM: M=32 batch rows per workgroup, K=512 (x[0:256] | h[256:512]),
// logical N = 4 gate groups x 256:  g0=r (both K phases), g1=z (both),
// g2=in_ (x phase only), g3=hn (h phase only).
// Each B fragment feeds two WMMAs (M sub-tiles) -> half the weight bandwidth
// per FLOP. Epilogue runs as two 16-row passes through a 64KB LDS exchange.
__global__ __launch_bounds__(256) void step_kernel(
    const unsigned short* __restrict__ plans_bf, const unsigned short* __restrict__ comm_bf,
    const unsigned short* __restrict__ wih_bf, const unsigned short* __restrict__ whh_bf,
    const float* __restrict__ b_ih, const float* __restrict__ b_hh,
    const unsigned char* __restrict__ row_has, const int* __restrict__ lenb,
    const unsigned char* __restrict__ order_valid,
    float* __restrict__ hbuf,              // (A,2,B,H) fp32, lives in d_out
    unsigned short* __restrict__ hbf,      // (A,2,B,H) bf16 mirror of h
    unsigned short* __restrict__ ybuf,     // (2,T,B,H) bf16, this agent
    int agent, int step) {
  __shared__ unsigned short Alds[32][520];   // 32 rows x 512 K (+8 pad), 33 KB
  __shared__ float ep[4][16][256];           // 64 KB gate exchange (per pass)
  const int mtile = blockIdx.x, dir = blockIdx.y;
  const int tid = threadIdx.x, lane = tid & 31, wave = tid >> 5;
  const int g = wave >> 1, nlo = (wave & 1) * 128;
  const int half = lane >> 4, col = lane & 15;
  const int r = dir ? (NT - 1 - step) : step;        // compact slot consumed

  // ---- cooperative A-panel staging: 8 threads per row, 64 bf16 per thread ----
  {
    const int row = tid >> 3, seg = tid & 7;
    const int bS = mtile * 32 + row;
    const bool vS = (r < lenb[bS]);
    const int toS = order_valid[r * NB + bS] & 7;
    const unsigned short* pR = plans_bf + ((size_t)bS * NA + agent) * NP;
    const unsigned short* cB = row_has[bS] ? comm_bf : plans_bf;
    const unsigned short* cR = cB + ((size_t)bS * NA + toS) * NP;
    const unsigned short* hR = hbf + (((size_t)agent * 2 + dir) * NB + bS) * NH;
    const int k = seg * 32;                 // 0..224
    const unsigned short* xs = (k < NP) ? (pR + k) : (cR + (k - NP));
    uint4 x0 = *(const uint4*)(xs);
    uint4 x1 = *(const uint4*)(xs + 8);
    uint4 x2 = *(const uint4*)(xs + 16);
    uint4 x3 = *(const uint4*)(xs + 24);
    uint4 h0 = *(const uint4*)(hR + k);
    uint4 h1 = *(const uint4*)(hR + k + 8);
    uint4 h2 = *(const uint4*)(hR + k + 16);
    uint4 h3 = *(const uint4*)(hR + k + 24);
    if (!vS) {                              // compacted-out row -> zeros
      x0 = x1 = x2 = x3 = h0 = h1 = h2 = h3 = uint4{0u, 0u, 0u, 0u};
    }
    *(uint4*)&Alds[row][k]            = x0;
    *(uint4*)&Alds[row][k + 8]        = x1;
    *(uint4*)&Alds[row][k + 16]       = x2;
    *(uint4*)&Alds[row][k + 24]       = x3;
    *(uint4*)&Alds[row][256 + k]      = h0;
    *(uint4*)&Alds[row][256 + k + 8]  = h1;
    *(uint4*)&Alds[row][256 + k + 16] = h2;
    *(uint4*)&Alds[row][256 + k + 24] = h3;
  }
  __syncthreads();

  const size_t wOff = (size_t)(agent * 2 + dir) * 768 * 256;
  const unsigned short* Wi = wih_bf + wOff;
  const unsigned short* Wh = whh_bf + wOff;

  v8f acc[2][8] = {};
  const int cBeg = (g == 3) ? 8 : 0;
  const int cEnd = (g == 2) ? 8 : 16;
  for (int c = cBeg; c < cEnd; ++c) {
    FragA af0, af1;
    const int k0 = c * 32 + half * 8;       // K runs [k0,k0+8) and [k0+16,k0+24)
    af0.q[0] = *(const uint4*)&Alds[col][k0];
    af0.q[1] = *(const uint4*)&Alds[col][k0 + 16];
    af1.q[0] = *(const uint4*)&Alds[16 + col][k0];
    af1.q[1] = *(const uint4*)&Alds[16 + col][k0 + 16];
    const bool ph1 = (c >= 8);
    const unsigned short* Wsrc = ph1 ? Wh : Wi;
    const int klocal = (ph1 ? (c - 8) : c) * 32 + half * 8;    // within 256
    #pragma unroll
    for (int tt = 0; tt < 8; ++tt) {
      const int j = nlo + tt * 16 + col;                        // 0..255 in gate
      const int rowIdx = (g == 0) ? j : (g == 1) ? (256 + j) : (512 + j);
      const unsigned short* wp = Wsrc + (size_t)rowIdx * 256 + klocal;
      FragA bf_;
      bf_.q[0] = *(const uint4*)wp;
      bf_.q[1] = *(const uint4*)(wp + 16);
      acc[0][tt] = __builtin_amdgcn_wmma_f32_16x16x32_bf16(
          false, af0.v, false, bf_.v, (short)0, acc[0][tt], false, false);
      acc[1][tt] = __builtin_amdgcn_wmma_f32_16x16x32_bf16(
          false, af1.v, false, bf_.v, (short)0, acc[1][tt], false, false);
    }
  }

  // ---- two 16-row epilogue passes through the 64KB gate exchange ----
  const float* bi = b_ih + (size_t)(agent * 2 + dir) * 768;
  const float* bh = b_hh + (size_t)(agent * 2 + dir) * 768;
  #pragma unroll
  for (int sub = 0; sub < 2; ++sub) {
    __syncthreads();                         // ep free / previous pass done
    #pragma unroll
    for (int tt = 0; tt < 8; ++tt) {
      const int j = nlo + tt * 16 + col;
      #pragma unroll
      for (int rr = 0; rr < 8; ++rr) ep[g][rr + half * 8][j] = acc[sub][tt][rr];
    }
    __syncthreads();

    const int row = tid >> 4, j0 = (tid & 15) * 16;
    const int bb = mtile * 32 + sub * 16 + row;
    const bool v = (r < lenb[bb]);
    float* hR = hbuf + (((size_t)agent * 2 + dir) * NB + bb) * NH;
    unsigned short* hbR = hbf + (((size_t)agent * 2 + dir) * NB + bb) * NH;
    unsigned short* yR = ybuf + (((size_t)dir * NT + step) * NB + bb) * NH;
    #pragma unroll
    for (int jj = 0; jj < 16; ++jj) {
      const int j = j0 + jj;
      const float rg = ep[0][row][j] + bi[j] + bh[j];
      const float zg = ep[1][row][j] + bi[256 + j] + bh[256 + j];
      const float ig = ep[2][row][j] + bi[512 + j];
      const float hg = ep[3][row][j] + bh[512 + j];
      const float rs = 1.f / (1.f + __expf(-rg));
      const float zs = 1.f / (1.f + __expf(-zg));
      const float nn = tanhf(ig + rs * hg);
      const float hold = hR[j];
      const float hnew = (1.f - zs) * nn + zs * hold;
      const unsigned short hb = f2bf_u(hnew);
      if (v) { hR[j] = hnew; hbR[j] = hb; }
      yR[j] = v ? hb : (unsigned short)0;
    }
  }
}

// ---------------------------------------------------------------------------
// Kernel 4: per-agent coordinator MLP on masked_scatter-gathered scores.
// GEMM: M=T*B rows, N=256, K=512 (yf | yb reversed), all-bf16 operands.
__global__ __launch_bounds__(256) void mlp_kernel(
    const unsigned short* __restrict__ ybuf, const int* __restrict__ srcMap,
    const unsigned short* __restrict__ w1_bf, const float* __restrict__ b1,
    const float* __restrict__ w2, const float* __restrict__ b2,
    float* __restrict__ coord, int agent) {
  const int mtile = blockIdx.x;                 // 0..(NT*NB/16 - 1)
  const int tid = threadIdx.x, lane = tid & 31, wave = tid >> 5;
  const int half = lane >> 4, col = lane & 15;

  // Gathered A row: (t,b) -> score row K = srcMap (torch masked_scatter order)
  const int grow = mtile * 16 + col;            // flat t*NB+b
  const int srcK = srcMap[grow];
  const bool vs = srcK >= 0;
  const int t2  = vs ? (srcK >> 12) : 0;
  const int b2i = vs ? (srcK & (NB - 1)) : 0;
  const unsigned short* yf = ybuf + (((size_t)0 * NT + t2) * NB + b2i) * NH;
  const unsigned short* yb = ybuf + (((size_t)1 * NT + (NT - 1 - t2)) * NB + b2i) * NH;
  const unsigned short* W1 = w1_bf + (size_t)agent * 256 * 512;

  v8f acc[2] = {};
  for (int c = 0; c < 16; ++c) {
    FragA af;
    const int k0 = c * 32 + half * 8;
    const int k1 = k0 + 16;
    const unsigned short* p0 = (k0 < 256) ? (yf + k0) : (yb + (k0 - 256));
    const unsigned short* p1 = (k1 < 256) ? (yf + k1) : (yb + (k1 - 256));
    af.q[0] = *(const uint4*)p0;
    af.q[1] = *(const uint4*)p1;
    if (!vs) {
      af.q[0] = uint4{0u, 0u, 0u, 0u};
      af.q[1] = uint4{0u, 0u, 0u, 0u};
    }
    const int klocal = c * 32 + half * 8;
    #pragma unroll
    for (int tt = 0; tt < 2; ++tt) {
      const int n = wave * 32 + tt * 16 + col;
      const unsigned short* wp = W1 + (size_t)n * 512 + klocal;
      FragA bf_;
      bf_.q[0] = *(const uint4*)wp;
      bf_.q[1] = *(const uint4*)(wp + 16);
      acc[tt] = __builtin_amdgcn_wmma_f32_16x16x32_bf16(
          false, af.v, false, bf_.v, (short)0, acc[tt], false, false);
    }
  }

  __shared__ float ep[16][256];
  __shared__ float red[32][8];
  #pragma unroll
  for (int tt = 0; tt < 2; ++tt) {
    const int n = wave * 32 + tt * 16 + col;
    #pragma unroll
    for (int rr = 0; rr < 8; ++rr) ep[rr + half * 8][n] = acc[tt][rr];
  }
  __syncthreads();
  {
    const int row = tid >> 4, j0 = (tid & 15) * 16;
    #pragma unroll
    for (int jj = 0; jj < 16; ++jj) {
      const int j = j0 + jj;
      ep[row][j] = fmaxf(ep[row][j] + b1[agent * 256 + j], 0.f);
    }
  }
  __syncthreads();
  {
    const int pair = tid >> 3;                // (row,ch): 32 pairs
    const int sub = tid & 7;                  // 8 partial threads each
    const int row = pair >> 1, ch = pair & 1;
    const float* W2 = w2 + ((size_t)agent * 2 + ch) * 256;
    float s = 0.f;
    const int jb = sub * 32;
    for (int j = jb; j < jb + 32; ++j) s += ep[row][j] * W2[j];
    red[pair][sub] = s;
  }
  __syncthreads();
  if (tid < 32) {
    const int row = tid >> 1, ch = tid & 1;
    float s = b2[agent * 2 + ch];
    #pragma unroll
    for (int k = 0; k < 8; ++k) s += red[tid][k];
    const int gr = mtile * 16 + row;
    const int t = gr >> 12, bq = gr & (NB - 1);
    coord[(((size_t)agent * NT + t) * NB + bq) * 2 + ch] = s;
  }
}

// ---------------------------------------------------------------------------
extern "C" void kernel_launch(void* const* d_in, const int* in_sizes, int n_in,
                              void* d_out, int out_size, void* d_ws, size_t ws_size,
                              hipStream_t stream) {
  (void)in_sizes; (void)n_in; (void)out_size; (void)ws_size;
  const float* plans        = (const float*)d_in[0];
  const float* comm_plans   = (const float*)d_in[1];
  const float* glob_hiddens = (const float*)d_in[2];
  const float* gw_ih        = (const float*)d_in[3];
  const float* gw_hh        = (const float*)d_in[4];
  const float* gb_ih        = (const float*)d_in[5];
  const float* gb_hh        = (const float*)d_in[6];
  const float* mw1          = (const float*)d_in[7];
  const float* mb1          = (const float*)d_in[8];
  const float* mw2          = (const float*)d_in[9];
  const float* mb2          = (const float*)d_in[10];

  float* out = (float*)d_out;
  const size_t coordElems = (size_t)NA * NT * NB * 2;          // 524288
  float* hbuf = out + coordElems;                              // (A,2,B,H) f32

  // ---- workspace carve-up (~99 MB) ----
  char* ws = (char*)d_ws;
  size_t off = 0;
  unsigned short* ybuf = (unsigned short*)(ws + off); off += (size_t)2 * NT * NB * NH * 2;
  unsigned short* hbf  = (unsigned short*)(ws + off); off += (size_t)NA * 2 * NB * NH * 2;
  unsigned short* plans_bf = (unsigned short*)(ws + off); off += (size_t)NB * NA * NP * 2;
  unsigned short* comm_bf  = (unsigned short*)(ws + off); off += (size_t)NB * NA * NP * 2;
  unsigned short* wih_bf = (unsigned short*)(ws + off); off += (size_t)NA * 2 * 768 * 256 * 2;
  unsigned short* whh_bf = (unsigned short*)(ws + off); off += (size_t)NA * 2 * 768 * 256 * 2;
  unsigned short* w1_bf  = (unsigned short*)(ws + off); off += (size_t)NA * 256 * 512 * 2;
  int* srcMap = (int*)(ws + off);            off += (size_t)NT * NB * sizeof(int);
  int* lenb   = (int*)(ws + off);            off += (size_t)NB * sizeof(int);
  unsigned char* cp_nz = (unsigned char*)(ws + off); off += (size_t)NT * NB;
  unsigned char* pl_nz = (unsigned char*)(ws + off); off += (size_t)NT * NB;
  unsigned char* maskA = (unsigned char*)(ws + off); off += (size_t)NT * NB;
  unsigned char* row_has = (unsigned char*)(ws + off); off += (size_t)NB;
  unsigned char* order_valid = (unsigned char*)(ws + off); off += (size_t)NT * NB;

  // h state (f32) lives in the output region; re-seed every call
  hipMemcpyAsync(hbuf, glob_hiddens, (size_t)NA * 2 * NB * NH * sizeof(float),
                 hipMemcpyDeviceToDevice, stream);

  // ---- one-time bf16 prepack (per call; bandwidth-bound) ----
  auto cvt = [&](const float* src, unsigned short* dst, size_t n) {
    cvt_bf16_kernel<<<dim3((unsigned)((n + 1023) / 1024)), 256, 0, stream>>>(src, dst, (int)n);
  };
  cvt(plans,        plans_bf, (size_t)NB * NA * NP);
  cvt(comm_plans,   comm_bf,  (size_t)NB * NA * NP);
  cvt(glob_hiddens, hbf,      (size_t)NA * 2 * NB * NH);
  cvt(gw_ih,        wih_bf,   (size_t)NA * 2 * 768 * 256);
  cvt(gw_hh,        whh_bf,   (size_t)NA * 2 * 768 * 256);
  cvt(mw1,          w1_bf,    (size_t)NA * 256 * 512);

  mask_kernel<<<dim3((NT * NB) / 8), 256, 0, stream>>>(plans, comm_plans, cp_nz, pl_nz);
  scan_kernel<<<dim3(1), 1024, 0, stream>>>(cp_nz, pl_nz, maskA, row_has, lenb,
                                            order_valid, srcMap);

  for (int a = 0; a < NA; ++a) {
    for (int s = 0; s < NT; ++s) {
      step_kernel<<<dim3(NB / 32, 2), 256, 0, stream>>>(
          plans_bf, comm_bf, wih_bf, whh_bf, gb_ih, gb_hh,
          row_has, lenb, order_valid, hbuf, hbf, ybuf, a, s);
    }
    mlp_kernel<<<dim3((NT * NB) / 16), 256, 0, stream>>>(
        ybuf, srcMap, w1_bf, mb1, mw2, mb2, out, a);
  }
}
